// MambaBlock_18021682774266
// MI455X (gfx1250) — compile-verified, matrix operand order checked by reading, not run on hardware
//
#include <hip/hip_runtime.h>
#include <hip/hip_bf16.h>

// ---------------- types ----------------
typedef __attribute__((ext_vector_type(16))) __bf16        v16bf;
typedef __attribute__((ext_vector_type(8)))  float         v8f;
typedef __attribute__((ext_vector_type(8)))  unsigned int  v8u;
typedef __attribute__((ext_vector_type(4)))  unsigned int  v4u;

#define TM 64
#define TN 128
#define TK 64
#define LSTR (TK + 8)   // 72 bf16 = 144 B row stride (multiple of 16 B)

// ---------------------------------------------------------------------------
// CDNA5 async copy: global -> LDS, 16 B per lane, tracked by ASYNCcnt.
// LDS byte offset = low 32 bits of the generic pointer to a __shared__ object.
// ---------------------------------------------------------------------------
__device__ __forceinline__ void async_cp16(const __bf16* g, __bf16* l) {
    unsigned loff = (unsigned)(unsigned long long)l;          // LDS byte offset
    unsigned long long ga = (unsigned long long)g;
    asm volatile("global_load_async_to_lds_b128 %0, %1, off"
                 :: "v"(loff), "v"(ga) : "memory");
}
__device__ __forceinline__ void wait_async0() {
#if __has_builtin(__builtin_amdgcn_s_wait_asynccnt)
    __builtin_amdgcn_s_wait_asynccnt(0);
#else
    asm volatile("s_wait_asynccnt 0x0" ::: "memory");
#endif
}

// ---------------------------------------------------------------------------
// bf16 WMMA GEMM, async-staged + double-buffered:
//   C[M,N] = A[M,K] @ B[K,N],  A is bf16 [M][K],  Bt is bf16 [N][K] (pre-T,
//   N padded to TN so tile loads never need guards).
// MODE 0: plain store.  MODE 1: softplus(acc + bias[col]).
// MODE 2: plain store + bf16 aux copy of cols < 64 (xproj -> dt input).
// NGUARD: bounds-check N on the C store only (xproj GEMM, N=96).
// Block: 128 thr = 4 waves; wave tile 32x64 = 2x4 wmma frags.
// ---------------------------------------------------------------------------
template<int MODE, bool NGUARD>
__global__ __launch_bounds__(128)
void mamba_gemm_bf16_wmma(const __bf16* __restrict__ A, int lda,
                          const __bf16* __restrict__ Bt, int ldb,
                          float* __restrict__ C, int ldc,
                          int M, int N, int K,
                          const float* __restrict__ bias,
                          __bf16* __restrict__ aux)
{
    __shared__ __bf16 As[2][TM][LSTR];
    __shared__ __bf16 Bs[2][TN][LSTR];

    const int tid  = threadIdx.x;
    const int row0 = blockIdx.y * TM;
    const int col0 = blockIdx.x * TN;

    const int wave  = tid >> 5;          // 4 waves (wave32)
    const int lane  = tid & 31;
    const int lr    = lane & 15;
    const int hi    = lane >> 4;
    const int kbase = hi * 8;            // 16-bit fragment K offset per ISA layout
    const int m_off = (wave >> 1) * 32;  // wave tile: 32 rows x 64 cols
    const int n_off = (wave & 1) * 64;

    v8f acc[2][4] = {};

    // ---- async tile stage: A 64x64 (8 chunks/row), B 128x64 ----
    auto stage = [&](int buf, int k0) {
        #pragma unroll
        for (int i = 0; i < 4; ++i) {
            int q = tid + i * 128;
            int r = q >> 3, c = q & 7;
            async_cp16(&A[(long)(row0 + r) * lda + k0 + c * 8], &As[buf][r][c * 8]);
        }
        #pragma unroll
        for (int i = 0; i < 8; ++i) {
            int q = tid + i * 128;
            int r = q >> 3, c = q & 7;
            async_cp16(&Bt[(long)(col0 + r) * ldb + k0 + c * 8], &Bs[buf][r][c * 8]);
        }
    };

    stage(0, 0);
    wait_async0();
    __syncthreads();

    int buf = 0;
    for (int k0 = 0; k0 < K; k0 += TK) {
        const int kn = k0 + TK;
        if (kn < K) stage(buf ^ 1, kn);          // prefetch next tile (async)

        #pragma unroll
        for (int kk = 0; kk < TK; kk += 32) {
            v16bf af[2], bfr[4];
            #pragma unroll
            for (int i = 0; i < 2; ++i) {
                const __bf16* rp = &As[buf][m_off + i * 16 + lr][0];
                v4u lo  = *(const v4u*)&rp[kk + kbase];
                v4u hi4 = *(const v4u*)&rp[kk + 16 + kbase];
                v8u u = { lo[0], lo[1], lo[2], lo[3], hi4[0], hi4[1], hi4[2], hi4[3] };
                af[i] = __builtin_bit_cast(v16bf, u);
            }
            #pragma unroll
            for (int j = 0; j < 4; ++j) {
                const __bf16* rp = &Bs[buf][n_off + j * 16 + lr][0];
                v4u lo  = *(const v4u*)&rp[kk + kbase];
                v4u hi4 = *(const v4u*)&rp[kk + 16 + kbase];
                v8u u = { lo[0], lo[1], lo[2], lo[3], hi4[0], hi4[1], hi4[2], hi4[3] };
                bfr[j] = __builtin_bit_cast(v16bf, u);
            }
            #pragma unroll
            for (int i = 0; i < 2; ++i)
                #pragma unroll
                for (int j = 0; j < 4; ++j)
                    acc[i][j] = __builtin_amdgcn_wmma_f32_16x16x32_bf16(
                        false, af[i], false, bfr[j],
                        (short)0, acc[i][j], false, false);
        }

        if (kn < K) wait_async0();               // own async loads done
        __syncthreads();                         // everyone's loads/reads done
        buf ^= 1;
    }

    // ---- store: f32 C/D layout -> M = v + 8*hi, N = lr ----
    #pragma unroll
    for (int i = 0; i < 2; ++i)
        #pragma unroll
        for (int j = 0; j < 4; ++j) {
            int gc = col0 + n_off + j * 16 + lr;
            if (!NGUARD || gc < N) {
                #pragma unroll
                for (int v = 0; v < 8; ++v) {
                    int gr = row0 + m_off + i * 16 + hi * 8 + v;
                    float val = acc[i][j][v];
                    if (MODE == 1) {
                        val += bias[gc];
                        val = (val > 20.0f) ? val : log1pf(__expf(val));
                    }
                    C[(long)gr * ldc + gc] = val;
                    if (MODE == 2 && gc < 64)
                        aux[(long)gr * 64 + gc] = (__bf16)val;
                }
            }
        }
}

// ---------------------------------------------------------------------------
// f32 -> bf16 elementwise (row-major copy)
// ---------------------------------------------------------------------------
__global__ __launch_bounds__(256)
void cvt_bf16(const float* __restrict__ in, __bf16* __restrict__ out, long n)
{
    long i = (long)blockIdx.x * 256 + threadIdx.x;
    if (i < n) out[i] = (__bf16)in[i];
}

// f32 [K][N] -> bf16 transposed [rows_out][K], rows >= N zero-padded.
__global__ __launch_bounds__(256)
void cvt_bf16_t(const float* __restrict__ in, int K, int N,
                __bf16* __restrict__ out, int rows_out)
{
    long i = (long)blockIdx.x * 256 + threadIdx.x;
    if (i >= (long)rows_out * K) return;
    int n = (int)(i / K), k = (int)(i % K);
    out[i] = (__bf16)((n < N) ? in[(long)k * N + n] : 0.0f);
}

// ---------------------------------------------------------------------------
// Causal depthwise conv (K=4) + SiLU.  x = xz[:, 0:2048] (row stride 4096).
// Writes f32 (scan input) and bf16 (GEMM input) copies.
// ---------------------------------------------------------------------------
__global__ __launch_bounds__(256)
void mamba_conv_silu(const float* __restrict__ xz,
                     const float* __restrict__ conv_w,
                     const float* __restrict__ conv_b,
                     float* __restrict__ xc, __bf16* __restrict__ xcb)
{
    long idx = (long)blockIdx.x * 256 + threadIdx.x;   // over 2048*2048
    int d = (int)(idx & 2047);
    int t = (int)(idx >> 11);
    int b = t >> 10, l = t & 1023;

    float acc = conv_b[d];
    #pragma unroll
    for (int k = 0; k < 4; ++k) {
        int ls = l - 3 + k;
        if (ls >= 0)
            acc += xz[((long)(b * 1024 + ls)) * 4096 + d] * conv_w[k * 2048 + d];
    }
    float s = acc / (1.0f + __expf(-acc));             // silu
    xc[idx]  = s;
    xcb[idx] = (__bf16)s;
}

// ---------------------------------------------------------------------------
// Selective scan (f32, recurrence).  16-lane group per (batch, channel d).
// h_n <- exp(dt*A[d,n])*h_n + dt*B[t,n]*x ;  y = sum_n h_n*C[t,n]
// yg (bf16) = (y + x*D[d]) * silu(z)  -> feeds the output GEMM directly.
// ---------------------------------------------------------------------------
__global__ __launch_bounds__(256)
void mamba_scan(const float* __restrict__ xc,
                const float* __restrict__ xz,
                const float* __restrict__ xproj,
                const float* __restrict__ dtf,
                const float* __restrict__ A_log,
                const float* __restrict__ Dv,
                __bf16* __restrict__ yg)
{
    int gt = blockIdx.x * 256 + threadIdx.x;   // 65536 total
    int n  = gt & 15;
    int g  = gt >> 4;                          // (b,d)
    int b  = g >> 11;
    int d  = g & 2047;

    float Ad = -__expf(A_log[d * 16 + n]);
    float Dd = Dv[d];
    float h  = 0.0f;

    for (int l = 0; l < 1024; ++l) {
        int t = b * 1024 + l;
        float dt = dtf[(long)t * 2048 + d];
        float xv = xc [(long)t * 2048 + d];
        float Bn = xproj[t * 96 + 64 + n];
        float Cn = xproj[t * 96 + 80 + n];

        h = __expf(dt * Ad) * h + dt * Bn * xv;

        float s = h * Cn;
        s += __shfl_xor(s, 1, 16);
        s += __shfl_xor(s, 2, 16);
        s += __shfl_xor(s, 4, 16);
        s += __shfl_xor(s, 8, 16);

        if (n == 0) {
            float y  = s + xv * Dd;
            float zv = xz[(long)t * 4096 + 2048 + d];
            float gz = zv / (1.0f + __expf(-zv));
            yg[(long)t * 2048 + d] = (__bf16)(y * gz);
        }
    }
}

// ---------------------------------------------------------------------------
extern "C" void kernel_launch(void* const* d_in, const int* in_sizes, int n_in,
                              void* d_out, int out_size, void* d_ws, size_t ws_size,
                              hipStream_t stream)
{
    const float* hs      = (const float*)d_in[0];  // [2,1024,1024]
    const float* W_in    = (const float*)d_in[1];  // [1024,4096]
    const float* conv_w  = (const float*)d_in[2];  // [4,2048]
    const float* conv_b  = (const float*)d_in[3];  // [2048]
    const float* W_xproj = (const float*)d_in[4];  // [2048,96]
    const float* W_dt    = (const float*)d_in[5];  // [64,2048]
    const float* b_dt    = (const float*)d_in[6];  // [2048]
    const float* A_log   = (const float*)d_in[7];  // [2048,16]
    const float* Dv      = (const float*)d_in[8];  // [2048]
    const float* W_out   = (const float*)d_in[9];  // [2048,1024]
    float* out = (float*)d_out;                    // [2,1024,1024]

    // ---- workspace layout (256 B aligned blocks) ----
    char* w = (char*)d_ws;
    auto alloc = [&](size_t bytes) { char* p = w; w += (bytes + 255) & ~(size_t)255; return p; };
    float*  xz     = (float*) alloc(2048L * 4096 * 4);  // xz (f32)
    float*  xc     = (float*) alloc(2048L * 2048 * 4);  // conv+silu (f32, scan)
    float*  xproj  = (float*) alloc(2048L * 96   * 4);  // xproj (f32, scan B/C)
    float*  dtf    = (float*) alloc(2048L * 2048 * 4);  // softplus dt (f32)
    __bf16* hs_bf  = (__bf16*)alloc(2048L * 1024 * 2);  // A of GEMM1
    __bf16* Wt_in  = (__bf16*)alloc(4096L * 1024 * 2);  // W_in^T
    __bf16* Wt_xp  = (__bf16*)alloc(128L  * 2048 * 2);  // W_xproj^T (pad 96->128)
    __bf16* Wt_dt  = (__bf16*)alloc(2048L * 64   * 2);  // W_dt^T
    __bf16* Wt_out = (__bf16*)alloc(1024L * 2048 * 2);  // W_out^T
    __bf16* xc_bf  = (__bf16*)alloc(2048L * 2048 * 2);  // A of GEMM3
    __bf16* xprojb = (__bf16*)alloc(2048L * 64   * 2);  // A of GEMM4 (dt cols)
    __bf16* yg_bf  = (__bf16*)alloc(2048L * 2048 * 2);  // A of GEMM6

    // ---- 0) one-pass bf16 conversions (weights transposed + padded) ----
    cvt_bf16  <<<(2048L * 1024 + 255) / 256, 256, 0, stream>>>(hs, hs_bf, 2048L * 1024);
    cvt_bf16_t<<<(4096L * 1024 + 255) / 256, 256, 0, stream>>>(W_in,    1024, 4096, Wt_in,  4096);
    cvt_bf16_t<<<(128L  * 2048 + 255) / 256, 256, 0, stream>>>(W_xproj, 2048, 96,   Wt_xp,  128);
    cvt_bf16_t<<<(2048L * 64   + 255) / 256, 256, 0, stream>>>(W_dt,    64,   2048, Wt_dt,  2048);
    cvt_bf16_t<<<(1024L * 2048 + 255) / 256, 256, 0, stream>>>(W_out,   2048, 1024, Wt_out, 1024);

    // ---- 1) xz = hs @ W_in        M=2048 N=4096 K=1024 ----
    mamba_gemm_bf16_wmma<0, false><<<dim3(4096 / TN, 2048 / TM), 128, 0, stream>>>(
        hs_bf, 1024, Wt_in, 1024, xz, 4096, 2048, 4096, 1024, nullptr, nullptr);

    // ---- 2) causal conv + silu -> xc (f32) + xc_bf ----
    mamba_conv_silu<<<(2048L * 2048) / 256, 256, 0, stream>>>(xz, conv_w, conv_b, xc, xc_bf);

    // ---- 3) xproj = xc @ W_xproj  M=2048 N=96 K=2048 (+bf16 dt-cols aux) ----
    mamba_gemm_bf16_wmma<2, true><<<dim3(1, 2048 / TM), 128, 0, stream>>>(
        xc_bf, 2048, Wt_xp, 2048, xproj, 96, 2048, 96, 2048, nullptr, xprojb);

    // ---- 4) dtf = softplus(xproj[:,:64] @ W_dt + b_dt)  M=2048 N=2048 K=64 ----
    mamba_gemm_bf16_wmma<1, false><<<dim3(2048 / TN, 2048 / TM), 128, 0, stream>>>(
        xprojb, 64, Wt_dt, 64, dtf, 2048, 2048, 2048, 64, b_dt, nullptr);

    // ---- 5) selective scan + gating -> yg_bf ----
    mamba_scan<<<65536 / 256, 256, 0, stream>>>(xc, xz, xproj, dtf, A_log, Dv, yg_bf);

    // ---- 6) out = yg @ W_out      M=2048 N=1024 K=2048 ----
    mamba_gemm_bf16_wmma<0, false><<<dim3(1024 / TN, 2048 / TM), 128, 0, stream>>>(
        yg_bf, 2048, Wt_out, 2048, out, 1024, 2048, 1024, 2048, nullptr, nullptr);
}
